// RNN_37074157699185
// MI455X (gfx1250) — matrix operand order repaired and checked by reading, not compiled, not run on hardware
//
#include <hip/hip_runtime.h>
#include <hip/hip_bf16.h>
#include <math.h>

#define T_DIM 8192
#define NS 1024
#define NH 512
#define NO 512
#define NWG_SCAN 16
#define HID_PER_WG (NH / NWG_SCAN)      // 32 hidden units per workgroup
#define ROWS_PER_WG (3 * HID_PER_WG)    // 96 W_hh rows per workgroup

// GEMM macro-tile
#define BM 64
#define BN 64
#define BK 32
#define BKP (BK + 4)                    // padded LDS stride (bank de-conflict)

typedef __attribute__((ext_vector_type(16))) __bf16 v16bf;
typedef __attribute__((ext_vector_type(8)))  float  v8f;
typedef int v4i __attribute__((vector_size(4 * sizeof(int))));

// gfx1250 async global->LDS path (ASYNCcnt), guarded so either toolchain/pass compiles.
#if defined(__AMDGCN__) && __has_builtin(__builtin_amdgcn_global_load_async_to_lds_b128) && \
    __has_builtin(__builtin_amdgcn_s_wait_asynccnt)
#define USE_ASYNC 1
typedef __attribute__((address_space(1))) v4i v4i_as1;
typedef __attribute__((address_space(3))) v4i v4i_as3;
#else
#define USE_ASYNC 0
#endif

__device__ inline void wait_stage() {
#if USE_ASYNC
  __builtin_amdgcn_s_wait_asynccnt(0);
#endif
}

// Stage a 64-row x 32-col f32 tile (row stride ldg in f32) into LDS (row stride BKP).
// 512 float4 transfers; 256 threads x 2.
__device__ inline void stage_tile(const float* __restrict__ g, int ldg,
                                  float* l, int tid) {
#pragma unroll
  for (int i = 0; i < 2; ++i) {
    const int idx = tid + i * 256;
    const int row = idx >> 3;
    const int c4  = (idx & 7) << 2;
#if USE_ASYNC
    __builtin_amdgcn_global_load_async_to_lds_b128(
        (v4i_as1*)(g + (size_t)row * ldg + c4),
        (v4i_as3*)(l + row * BKP + c4),
        0, 0);
#else
    *reinterpret_cast<float4*>(l + row * BKP + c4) =
        *reinterpret_cast<const float4*>(g + (size_t)row * ldg + c4);
#endif
  }
}

__device__ inline v16bf pack16(float4 a, float4 b, float4 c, float4 d) {
  v16bf f;
  f[0]  = (__bf16)a.x; f[1]  = (__bf16)a.y; f[2]  = (__bf16)a.z; f[3]  = (__bf16)a.w;
  f[4]  = (__bf16)b.x; f[5]  = (__bf16)b.y; f[6]  = (__bf16)b.z; f[7]  = (__bf16)b.w;
  f[8]  = (__bf16)c.x; f[9]  = (__bf16)c.y; f[10] = (__bf16)c.z; f[11] = (__bf16)c.w;
  f[12] = (__bf16)d.x; f[13] = (__bf16)d.y; f[14] = (__bf16)d.z; f[15] = (__bf16)d.w;
  return f;
}

// A fragment (16x32, M x K) from LDS subtile base (row stride BKP).
// ISA layout: lane l, row m=l&15; halves 0-7 = K hi..hi+7, halves 8-15 = K 16+hi..,
// hi = 8*(l>>4).
__device__ inline v16bf fragA_lds(const float* a, int lane) {
  const int hi = (lane >> 4) << 3;
  const float* r = a + (lane & 15) * BKP;
  return pack16(*(const float4*)(r + hi),      *(const float4*)(r + hi + 4),
                *(const float4*)(r + 16 + hi), *(const float4*)(r + 20 + hi));
}

// B fragment (32x16, K x N) from LDS subtile base; we store B row-major by output
// column (row stride BKP). ISA layout: lane l holds column n=l&15; halves 0-7 = K hi..hi+7,
// halves 8-15 = K 16+hi.., hi = 8*(l>>4).
__device__ inline v16bf fragB_lds(const float* b, int lane) {
  const int hi = (lane >> 4) << 3;
  const float* r = b + (lane & 15) * BKP;
  return pack16(*(const float4*)(r + hi),      *(const float4*)(r + hi + 4),
                *(const float4*)(r + 16 + hi), *(const float4*)(r + 20 + hi));
}

// C[M,N] = act(A[M,K] @ B[N,K]^T + bias[N]); act: 0 = identity, 1 = tanh.
// Block = 256 threads (8 waves), 64x64 macro-tile, double-buffered async LDS staging.
// Wave w: m-subtile (w&3), n-half (w>>2); computes two 16x16 tiles sharing one A frag.
__global__ __launch_bounds__(256) void gemm_bf16_wmma(
    const float* __restrict__ A, const float* __restrict__ Bw,
    const float* __restrict__ bias, float* __restrict__ C,
    int M, int N, int K, int act) {
  __shared__ float Asm[2][BM * BKP];
  __shared__ float Bsm[2][BN * BKP];

  const int tid  = threadIdx.x;
  const int lane = tid & 31;
  const int wave = tid >> 5;
  const int wm   = wave & 3;    // 0..3
  const int wn   = wave >> 2;   // 0..1
  const int m0   = blockIdx.y * BM;
  const int n0   = blockIdx.x * BN;

  const float* Ag = A  + (size_t)m0 * K;
  const float* Bg = Bw + (size_t)n0 * K;

  v8f acc0 = {}, acc1 = {};

  stage_tile(Ag, K, Asm[0], tid);
  stage_tile(Bg, K, Bsm[0], tid);

  const int nsteps = K / BK;
  for (int kt = 0; kt < nsteps; ++kt) {
    const int cur = kt & 1;
    wait_stage();
    __syncthreads();                       // buffer `cur` ready for all waves
    if (kt + 1 < nsteps) {                 // prefetch next tile into other buffer
      stage_tile(Ag + (kt + 1) * BK, K, Asm[cur ^ 1], tid);
      stage_tile(Bg + (kt + 1) * BK, K, Bsm[cur ^ 1], tid);
    }
    v16bf a  = fragA_lds(&Asm[cur][(wm * 16) * BKP], lane);
    v16bf b0 = fragB_lds(&Bsm[cur][(wn * 32 + 0)  * BKP], lane);
    v16bf b1 = fragB_lds(&Bsm[cur][(wn * 32 + 16) * BKP], lane);
    acc0 = __builtin_amdgcn_wmma_f32_16x16x32_bf16(false, a, false, b0, (short)0,
                                                   acc0, false, false);
    acc1 = __builtin_amdgcn_wmma_f32_16x16x32_bf16(false, a, false, b1, (short)0,
                                                   acc1, false, false);
    __syncthreads();                       // done reading `cur` before it is restaged
  }

  // C/D layout: lane l, VGPR v -> (m = v + 8*(l>>4), n = l&15)
  const int mo     = (lane >> 4) * 8;
  const int m_base = m0 + wm * 16 + mo;
  const int nA     = n0 + wn * 32 + (lane & 15);
  const int nB     = nA + 16;
  const float bvA = bias[nA], bvB = bias[nB];
#pragma unroll
  for (int v = 0; v < 8; ++v) {
    float r0 = acc0[v] + bvA;
    float r1 = acc1[v] + bvB;
    if (act == 1) { r0 = tanhf(r0); r1 = tanhf(r1); }
    C[(size_t)(m_base + v) * N + nA] = r0;
    C[(size_t)(m_base + v) * N + nB] = r1;
  }
}

// Re-initialize barrier counter and h state each launch (graph-replay safe).
__global__ void scan_init(unsigned* __restrict__ bar, float* __restrict__ hbuf,
                          const float* __restrict__ h0) {
  if (threadIdx.x == 0) *bar = 0u;
  for (int i = threadIdx.x; i < NH; i += blockDim.x)
    __hip_atomic_store(&hbuf[i], h0[i], __ATOMIC_RELAXED, __HIP_MEMORY_SCOPE_AGENT);
}

// Persistent GRU scan: 16 WGs, each owns 32 hidden units (96 W_hh rows in LDS).
// One device-wide barrier per timestep publishes hbuf[512] through L2 (agent scope).
__global__ __launch_bounds__(256) void gru_scan(
    const float* __restrict__ gi, const int* __restrict__ done,
    const float* __restrict__ W_hh, const float* __restrict__ b_hh,
    float* __restrict__ hbuf, float* __restrict__ Hnew,
    unsigned* __restrict__ bar) {
  extern __shared__ float smem[];
  float* Wsl   = smem;                            // ROWS_PER_WG * NH (192 KB)
  float* h_sh  = smem + ROWS_PER_WG * NH;         // NH
  float* gh_sh = h_sh + NH;                       // ROWS_PER_WG

  const int tid = threadIdx.x;
  const int wg  = blockIdx.x;

  // Stage this WG's W_hh slice: local row rr = g*32 + j <- global row g*NH + wg*32 + j
  for (int idx = tid; idx < ROWS_PER_WG * (NH / 4); idx += blockDim.x) {
    int rr = idx / (NH / 4);
    int c4 = idx % (NH / 4);
    int g = rr / HID_PER_WG, j = rr % HID_PER_WG;
    const float4* src =
        reinterpret_cast<const float4*>(W_hh + (size_t)(g * NH + wg * HID_PER_WG + j) * NH) + c4;
    reinterpret_cast<float4*>(Wsl + rr * NH)[c4] = *src;
  }
  __syncthreads();

  const int sub    = tid & 7;          // 8 lanes per dot product
  const int rowgrp = tid >> 3;         // 32 rows per pass
  const int swz    = (tid & 31) << 1;  // LDS bank-conflict swizzle

  for (int t = 0; t < T_DIM; ++t) {
    // Pull the freshly published h vector into LDS (DEV-scope loads, skip stale WGP$).
    h_sh[tid] = __hip_atomic_load(&hbuf[tid], __ATOMIC_RELAXED, __HIP_MEMORY_SCOPE_AGENT);
    h_sh[tid + 256] =
        __hip_atomic_load(&hbuf[tid + 256], __ATOMIC_RELAXED, __HIP_MEMORY_SCOPE_AGENT);
    __syncthreads();

    // gh[rr] = dot(W_hh_row, h) over K=512, split 8 ways then shuffle-reduced.
#pragma unroll
    for (int p = 0; p < 3; ++p) {
      const int rr = p * 32 + rowgrp;
      const float* wrow = Wsl + rr * NH;
      float acc = 0.f;
#pragma unroll 8
      for (int k = 0; k < 64; ++k) {
        const int koff = sub * 64 + ((k + swz) & 63);
        acc = fmaf(wrow[koff], h_sh[koff], acc);
      }
      acc += __shfl_xor(acc, 1);
      acc += __shfl_xor(acc, 2);
      acc += __shfl_xor(acc, 4);
      if (sub == 0) gh_sh[rr] = acc;
    }
    __syncthreads();

    // Gate math for this WG's 32 hidden units.
    if (tid < HID_PER_WG) {
      const int j  = tid;
      const int jg = wg * HID_PER_WG + j;
      const size_t gbase = (size_t)t * (3 * NH);
      const float gr  = gi[gbase + jg]          + gh_sh[j]      + b_hh[jg];
      const float gz  = gi[gbase + NH + jg]     + gh_sh[32 + j] + b_hh[NH + jg];
      const float gn  = gi[gbase + 2 * NH + jg];
      const float ghn = gh_sh[64 + j] + b_hh[2 * NH + jg];
      const float r  = 1.f / (1.f + __expf(-gr));
      const float z  = 1.f / (1.f + __expf(-gz));
      const float nn = tanhf(gn + r * ghn);
      const float hprev = h_sh[jg];
      const float hn = (1.f - z) * nn + z * hprev;
      Hnew[(size_t)t * NH + jg] = hn;                 // pre-reset h_new (feeds W2 GEMM)
      __hip_atomic_store(&hbuf[jg], hn * (1.f - (float)done[t]),
                         __ATOMIC_RELAXED, __HIP_MEMORY_SCOPE_AGENT);
    }
    __syncthreads();

    // Device-wide step barrier: monotonically increasing arrival counter.
    if (tid == 0) {
      __threadfence();
      __hip_atomic_fetch_add(bar, 1u, __ATOMIC_RELEASE, __HIP_MEMORY_SCOPE_AGENT);
      const unsigned target = (unsigned)(t + 1) * NWG_SCAN;
      while (__hip_atomic_load(bar, __ATOMIC_ACQUIRE, __HIP_MEMORY_SCOPE_AGENT) < target)
        __builtin_amdgcn_s_sleep(1);
      __threadfence();
    }
    __syncthreads();
  }
}

__global__ void copy_hfinal(const float* __restrict__ hbuf, float* __restrict__ dst) {
  const int i = threadIdx.x;
  if (i < NH)
    dst[i] = __hip_atomic_load(&hbuf[i], __ATOMIC_RELAXED, __HIP_MEMORY_SCOPE_AGENT);
}

extern "C" void kernel_launch(void* const* d_in, const int* in_sizes, int n_in,
                              void* d_out, int out_size, void* d_ws, size_t ws_size,
                              hipStream_t stream) {
  const float* obs  = (const float*)d_in[0];
  const int*   done = (const int*)  d_in[1];
  const float* h0   = (const float*)d_in[2];
  const float* W1   = (const float*)d_in[3];
  const float* b1   = (const float*)d_in[4];
  const float* W_ih = (const float*)d_in[5];
  const float* W_hh = (const float*)d_in[6];
  const float* b_ih = (const float*)d_in[7];
  const float* b_hh = (const float*)d_in[8];
  const float* W2   = (const float*)d_in[9];
  const float* b2   = (const float*)d_in[10];
  float* out = (float*)d_out;

  // Workspace layout.
  char* ws = (char*)d_ws;
  unsigned* bar = (unsigned*)ws;                       // barrier counter
  float* hbuf   = (float*)(ws + 256);                  // NH carried state
  float* x      = (float*)(ws + 4096);                 // [T, NH]
  float* gi     = x  + (size_t)T_DIM * NH;             // [T, 3NH]
  float* Hnew   = gi + (size_t)T_DIM * 3 * NH;         // [T, NH] pre-reset states

  scan_init<<<1, 256, 0, stream>>>(bar, hbuf, h0);

  const dim3 blk(256);
  // x = tanh(obs @ W1^T + b1)
  gemm_bf16_wmma<<<dim3(NH / BN, T_DIM / BM), blk, 0, stream>>>(
      obs, W1, b1, x, T_DIM, NH, NS, 1);
  // gi = x @ W_ih^T + b_ih
  gemm_bf16_wmma<<<dim3(3 * NH / BN, T_DIM / BM), blk, 0, stream>>>(
      x, W_ih, b_ih, gi, T_DIM, 3 * NH, NH, 0);

  // Sequential GRU scan (persistent, W_hh resident in LDS across 16 WGPs).
  const size_t lds = (size_t)(ROWS_PER_WG * NH + NH + ROWS_PER_WG) * sizeof(float);
  gru_scan<<<NWG_SCAN, 256, lds, stream>>>(gi, done, W_hh, b_hh, hbuf, Hnew, bar);

  // out = Hnew @ W2^T + b2
  gemm_bf16_wmma<<<dim3(NO / BN, T_DIM / BM), blk, 0, stream>>>(
      Hnew, W2, b2, out, T_DIM, NO, NH, 0);

  copy_hfinal<<<1, 512, 0, stream>>>(hbuf, out + (size_t)T_DIM * NO);
}
